// MoELayer_74371653698164
// MI455X (gfx1250) — matrix operand order repaired and checked by reading, not compiled, not run on hardware
//
#include <hip/hip_runtime.h>
#include <hip/hip_bf16.h>
#include <math.h>

typedef __bf16 bf16_t;
typedef __attribute__((ext_vector_type(16))) __bf16 v16bf;
typedef __attribute__((ext_vector_type(8)))  __bf16 v8bf;
typedef __attribute__((ext_vector_type(8)))  float  v8f;

#define N_TOK 8192
#define D_IN  1024
#define D_OUT 1024
#define NEXP  8
#define LDT   40            // LDS row pitch in bf16 elems: 80B, 16B-aligned, bank-friendly
#define TILE_ELEMS (128 * LDT)
#define STEPS (NEXP * (D_IN / 32))   // 256 flattened (expert, k-slab) steps

// ---------------- gate: p = softmax(x @ Wg + bg), [N, 8] ----------------
__global__ __launch_bounds__(256) void gate_kernel(
    const float* __restrict__ x, const float* __restrict__ Wg,
    const float* __restrict__ bg, float* __restrict__ p) {
  __shared__ float wg[D_IN * NEXP];
  const int tid = threadIdx.x;
  for (int i = tid; i < D_IN * NEXP; i += 256) wg[i] = Wg[i];
  __syncthreads();
  const int lane  = tid & 31;
  const int wave  = tid >> 5;
  const int token = blockIdx.x * 8 + wave;
  const float* xr = x + (size_t)token * D_IN;
  float acc[NEXP];
#pragma unroll
  for (int e = 0; e < NEXP; ++e) acc[e] = 0.f;
  for (int i = 0; i < D_IN / 32; ++i) {
    const int k = i * 32 + lane;
    const float xv = xr[k];
#pragma unroll
    for (int e = 0; e < NEXP; ++e) acc[e] += xv * wg[k * NEXP + e];
  }
#pragma unroll
  for (int e = 0; e < NEXP; ++e) {
#pragma unroll
    for (int off = 16; off > 0; off >>= 1)
      acc[e] += __shfl_xor(acc[e], off, 32);
  }
  float ex[NEXP], mx = -1e30f, s = 0.f;
#pragma unroll
  for (int e = 0; e < NEXP; ++e) { acc[e] += bg[e]; mx = fmaxf(mx, acc[e]); }
#pragma unroll
  for (int e = 0; e < NEXP; ++e) { ex[e] = __expf(acc[e] - mx); s += ex[e]; }
  const float inv = 1.f / s;
  if (lane == 0) {
#pragma unroll
    for (int e = 0; e < NEXP; ++e) p[(size_t)token * NEXP + e] = ex[e] * inv;
  }
}

// ------- transpose+convert We[e][i][o] (f32) -> Wt[e][o][i] (bf16) -------
__global__ __launch_bounds__(256) void wconv_kernel(
    const float* __restrict__ We, bf16_t* __restrict__ Wt) {
  __shared__ float tile[32][33];
  const int e  = blockIdx.z;
  const int i0 = blockIdx.y * 32;
  const int o0 = blockIdx.x * 32;
  const float* src = We + (size_t)e * D_IN * D_OUT;
  bf16_t*      dst = Wt + (size_t)e * D_IN * D_OUT;
  const int t  = threadIdx.x;
  const int oc = t & 31;
  const int ir = t >> 5;
#pragma unroll
  for (int s = 0; s < 4; ++s) {
    const int ii = ir + s * 8;
    tile[ii][oc] = src[(size_t)(i0 + ii) * D_OUT + (o0 + oc)];
  }
  __syncthreads();
  const int ic  = t & 31;
  const int orr = t >> 5;
#pragma unroll
  for (int s = 0; s < 4; ++s) {
    const int oo = orr + s * 8;
    dst[(size_t)(o0 + oo) * D_IN + (i0 + ic)] = (bf16_t)tile[ic][oo];
  }
}

// ---------- bias pre-fill: out[n,o] = sum_e p[n,e] * be[e,o] ----------
__global__ __launch_bounds__(256) void bias_kernel(
    const float* __restrict__ p, const float* __restrict__ be,
    float* __restrict__ out) {
  const int gid = blockIdx.x * 256 + threadIdx.x;
  const int n = gid >> 10;
  const int o = gid & (D_OUT - 1);
  const float* pn = p + (size_t)n * NEXP;
  float s = 0.f;
#pragma unroll
  for (int e = 0; e < NEXP; ++e) s += pn[e] * be[e * D_OUT + o];
  out[gid] = s;
}

// ---------------- fused expert GEMM (double-buffered pipeline) ----------------
struct Fetch {
  float4 f0, f1, f2, f3;   // 16 fp32 x values
  v8bf   b0, b1;           // 16 bf16 weight values
  float  pr;               // gate prob for this thread's row / expert
};

__device__ __forceinline__ void do_fetch(int step, const float* xrow,
                                         const bf16_t* bcol0,
                                         const float* p, int prow, Fetch& F) {
  const int e  = step >> 5;
  const int k0 = (step & 31) << 5;
  const float* xp = xrow + k0;
  F.f0 = *(const float4*)(xp + 0);
  F.f1 = *(const float4*)(xp + 4);
  F.f2 = *(const float4*)(xp + 8);
  F.f3 = *(const float4*)(xp + 12);
  const bf16_t* bp = bcol0 + (size_t)e * (D_IN * D_OUT) + k0;
  F.b0 = *(const v8bf*)(bp);
  F.b1 = *(const v8bf*)(bp + 8);
  F.pr = p[(size_t)prow * NEXP + e];
}

__device__ __forceinline__ void do_store(bf16_t* da, bf16_t* db, const Fetch& F) {
  const float pr = F.pr;
  v8bf a0, a1;
  a0[0] = (bf16_t)(F.f0.x * pr); a0[1] = (bf16_t)(F.f0.y * pr);
  a0[2] = (bf16_t)(F.f0.z * pr); a0[3] = (bf16_t)(F.f0.w * pr);
  a0[4] = (bf16_t)(F.f1.x * pr); a0[5] = (bf16_t)(F.f1.y * pr);
  a0[6] = (bf16_t)(F.f1.z * pr); a0[7] = (bf16_t)(F.f1.w * pr);
  a1[0] = (bf16_t)(F.f2.x * pr); a1[1] = (bf16_t)(F.f2.y * pr);
  a1[2] = (bf16_t)(F.f2.z * pr); a1[3] = (bf16_t)(F.f2.w * pr);
  a1[4] = (bf16_t)(F.f3.x * pr); a1[5] = (bf16_t)(F.f3.y * pr);
  a1[6] = (bf16_t)(F.f3.z * pr); a1[7] = (bf16_t)(F.f3.w * pr);
  *(v8bf*)(da)     = a0;
  *(v8bf*)(da + 8) = a1;
  *(v8bf*)(db)     = F.b0;
  *(v8bf*)(db + 8) = F.b1;
}

__device__ __forceinline__ void do_compute(const bf16_t* bufA, const bf16_t* bufB,
                                           int arow, int bcol, int klo, int kbB,
                                           v8f acc[2][4]) {
  // ISA fragment layouts (wave32):
  // A 16-bit 16x32: lanes 0-15 hold K in [0,8)+[16,24); lanes 16-31: [8,16)+[24,32)
  // B 16-bit 32x16: same K split as A: lanes 0-15 hold K [0,8)+[16,24); lanes 16-31: [8,16)+[24,32)
  v16bf afrag[2], bfrag[4];
#pragma unroll
  for (int mt = 0; mt < 2; ++mt) {
    const bf16_t* pa = bufA + (arow + mt * 16) * LDT + klo;
    v8bf lo = *(const v8bf*)(pa);
    v8bf hi = *(const v8bf*)(pa + 16);
    afrag[mt] = __builtin_shufflevector(
        lo, hi, 0, 1, 2, 3, 4, 5, 6, 7, 8, 9, 10, 11, 12, 13, 14, 15);
  }
#pragma unroll
  for (int nt = 0; nt < 4; ++nt) {
    const bf16_t* pb = bufB + (bcol + nt * 16) * LDT + kbB;
    v8bf lo = *(const v8bf*)(pb);
    v8bf hi = *(const v8bf*)(pb + 16);
    bfrag[nt] = __builtin_shufflevector(
        lo, hi, 0, 1, 2, 3, 4, 5, 6, 7, 8, 9, 10, 11, 12, 13, 14, 15);
  }
#pragma unroll
  for (int mt = 0; mt < 2; ++mt)
#pragma unroll
    for (int nt = 0; nt < 4; ++nt)
      acc[mt][nt] = __builtin_amdgcn_wmma_f32_16x16x32_bf16(
          false, afrag[mt], false, bfrag[nt], (short)0, acc[mt][nt],
          false, false);
}

__global__ __launch_bounds__(256) void moe_gemm_kernel(
    const float* __restrict__ x, const bf16_t* __restrict__ Wt,
    const float* __restrict__ p, float* __restrict__ out) {
  __shared__ bf16_t ldsA[2 * TILE_ELEMS];
  __shared__ bf16_t ldsB[2 * TILE_ELEMS];
  const int tid  = threadIdx.x;
  const int lane = tid & 31;
  const int wave = tid >> 5;
  const int wm = wave >> 1;            // 0..3 : 32-row strip
  const int wn = wave & 1;             // 0..1 : 64-col strip
  const int row0 = blockIdx.y * 128;   // token tile
  const int col0 = blockIdx.x * 128;   // d_out tile

  const int frow = tid >> 1;           // 0..127 cooperative-fill row
  const int fseg = tid & 1;            // which 16-elem segment of BK=32
  const int prow = row0 + frow;
  const float*  xrow  = x  + (size_t)prow * D_IN + fseg * 16;
  const bf16_t* bcol0 = Wt + (size_t)(col0 + frow) * D_IN + fseg * 16;
  bf16_t* const daBase = &ldsA[frow * LDT + fseg * 16];
  bf16_t* const dbBase = &ldsB[frow * LDT + fseg * 16];

  v8f zero = {};
  v8f acc[2][4];
#pragma unroll
  for (int i = 0; i < 2; ++i)
#pragma unroll
    for (int j = 0; j < 4; ++j) acc[i][j] = zero;

  const int klo  = (lane >> 4) * 8;
  const int kbB  = (lane >> 4) * 8;
  const int arow = wm * 32 + (lane & 15);
  const int bcol = wn * 64 + (lane & 15);

  // ---- pipeline prologue: stage step 0 into buffer 0 ----
  Fetch F;
  do_fetch(0, xrow, bcol0, p, prow, F);
  do_store(daBase, dbBase, F);
  __syncthreads();

  // ---- main pipeline: fetch(i+1) || compute(i) -> store(i+1) -> barrier ----
  for (int i = 0; i < STEPS - 1; ++i) {
    const int cur = i & 1;
    Fetch Fn;
    do_fetch(i + 1, xrow, bcol0, p, prow, Fn);
    do_compute(&ldsA[cur * TILE_ELEMS], &ldsB[cur * TILE_ELEMS],
               arow, bcol, klo, kbB, acc);
    const int nxt = cur ^ 1;
    do_store(daBase + nxt * TILE_ELEMS, dbBase + nxt * TILE_ELEMS, Fn);
    __syncthreads();
  }
  // ---- epilogue step ----
  {
    const int cur = (STEPS - 1) & 1;
    do_compute(&ldsA[cur * TILE_ELEMS], &ldsB[cur * TILE_ELEMS],
               arow, bcol, klo, kbB, acc);
  }

  // ---- epilogue: out was pre-filled with gate-weighted bias; add GEMM ----
  // C layout: VGPR r -> M = r (+8 for lanes 16-31), N = lane & 15
  const int mbase = row0 + wm * 32 + ((lane >> 4) << 3);
  const int nbase = col0 + wn * 64 + (lane & 15);
#pragma unroll
  for (int mt = 0; mt < 2; ++mt)
#pragma unroll
    for (int nt = 0; nt < 4; ++nt)
#pragma unroll
      for (int r = 0; r < 8; ++r) {
        const size_t idx =
            (size_t)(mbase + mt * 16 + r) * D_OUT + (nbase + nt * 16);
        out[idx] += acc[mt][nt][r];
      }
}

extern "C" void kernel_launch(void* const* d_in, const int* in_sizes, int n_in,
                              void* d_out, int out_size, void* d_ws, size_t ws_size,
                              hipStream_t stream) {
  const float* x  = (const float*)d_in[0];
  const float* We = (const float*)d_in[1];
  const float* be = (const float*)d_in[2];
  const float* Wg = (const float*)d_in[3];
  const float* bg = (const float*)d_in[4];
  float* out = (float*)d_out;

  // workspace: [ Wt bf16: E*D_IN*D_OUT ][ p f32: N*E ]
  bf16_t* Wt = (bf16_t*)d_ws;
  float*  p  = (float*)((char*)d_ws +
                        (size_t)NEXP * D_IN * D_OUT * sizeof(bf16_t));

  gate_kernel<<<N_TOK / 8, 256, 0, stream>>>(x, Wg, bg, p);
  wconv_kernel<<<dim3(D_OUT / 32, D_IN / 32, NEXP), 256, 0, stream>>>(We, Wt);
  bias_kernel<<<(N_TOK * D_OUT) / 256, 256, 0, stream>>>(p, be, out);
  moe_gemm_kernel<<<dim3(D_OUT / 128, N_TOK / 128), 256, 0, stream>>>(x, Wt, p, out);
}
